// EPEG_60155311947834
// MI455X (gfx1250) — matrix operand order repaired and checked by reading, not compiled, not run on hardware
//
#include <hip/hip_runtime.h>

// ---------------------------------------------------------------------------
// Fused windowed attention for MI455X (gfx1250), wave32 + WMMA bf16.
// B=4, L2=256, M=8, D=512, H=8 -> N=1024 windows, SL=64 tokens, DK=64.
// One workgroup (256 thr = 8 waves) per window; wave w owns head w.
// ---------------------------------------------------------------------------

typedef __bf16 bf16_t;
typedef bf16_t v16bf __attribute__((ext_vector_type(16)));
typedef bf16_t v8bf  __attribute__((ext_vector_type(8)));
typedef bf16_t v2bf  __attribute__((ext_vector_type(2)));
typedef float  v8f   __attribute__((ext_vector_type(8)));

#define NWIN   1024
#define SL     64
#define DMODEL 512
#define NHEAD  8
#define DK     64

// LDS layout (bytes), total 320 KB:
//   [0      , 65536) Qs : per-head bf16 [h][64][64]
//   [65536  ,131072) Ks : per-head bf16 [h][64][64]
//   [131072 ,196608) Vt : per-head bf16 [h][dk][64]  (transposed)
//   [196608 ,327680) S  : phase1: X bf16 64x512 (first 64K)
//                         phase2: per-wave 16KB f32 scores; attn bf16 in
//                         first 8KB, O bf16 in second 8KB.
#define LDS_Q   0
#define LDS_K   65536
#define LDS_VT  131072
#define LDS_S   196608
#define LDS_BYTES 327680

// A-fragment (16x32 bf16): lane m=l%16, khalf=l/16:
//   a[0..7]  = M[m][k0 + khalf*8 + 0..7]
//   a[8..15] = M[m][k0 + 16 + khalf*8 + 0..7]
__device__ __forceinline__ v16bf load_afrag(const bf16_t* __restrict__ base,
                                            int row_stride, int m0, int k0,
                                            int lane) {
  const int mm = lane & 15, kh = lane >> 4;
  const v8bf lo = *(const v8bf*)(base + (m0 + mm) * row_stride + k0 + kh * 8);
  const v8bf hi = *(const v8bf*)(base + (m0 + mm) * row_stride + k0 + 16 + kh * 8);
  return __builtin_shufflevector(lo, hi, 0, 1, 2, 3, 4, 5, 6, 7,
                                 8, 9, 10, 11, 12, 13, 14, 15);
}

// B-fragment (32x16 bf16), B[k][n] taken from a row-major matrix R[n][k]:
// lane n=l%16, khalf=l/16: b[0..7] = R[n0+n][k0 + khalf*8 + 0..7],
//   b[8..15] = R[n0+n][k0 + 16 + khalf*8 + 0..7]
__device__ __forceinline__ v16bf load_bfrag(const bf16_t* __restrict__ base,
                                            int row_stride, int n0, int k0,
                                            int lane) {
  const int nn = lane & 15, kh = lane >> 4;
  const v8bf lo = *(const v8bf*)(base + (n0 + nn) * row_stride + k0 + kh * 8);
  const v8bf hi = *(const v8bf*)(base + (n0 + nn) * row_stride + k0 + 16 + kh * 8);
  return __builtin_shufflevector(lo, hi, 0, 1, 2, 3, 4, 5, 6, 7,
                                 8, 9, 10, 11, 12, 13, 14, 15);
}

__device__ __forceinline__ v8f wmma_bf16(v16bf a, v16bf b, v8f c) {
  return __builtin_amdgcn_wmma_f32_16x16x32_bf16(false, a, false, b,
                                                 (short)0, c, false, false);
}

// ---------------------------------------------------------------------------
// Kernel 0: convert the four 512x512 f32 weight matrices to bf16 in d_ws
// (order: Wq | Wk | Wv | Wo), so the fused kernel streams bf16 from L2.
// ---------------------------------------------------------------------------
__global__ __launch_bounds__(256) void weights_to_bf16(
    const float* __restrict__ Wq, const float* __restrict__ Wk,
    const float* __restrict__ Wv, const float* __restrict__ Wo,
    bf16_t* __restrict__ dst) {
  const int idx = blockIdx.x * 256 + threadIdx.x;   // 4*512*512 total
  const int m = idx >> 18;
  const int off = idx & 0x3FFFF;
  float v;
  if (m == 0)      v = Wq[off];
  else if (m == 1) v = Wk[off];
  else if (m == 2) v = Wv[off];
  else             v = Wo[off];
  dst[idx] = (bf16_t)v;
}

// ---------------------------------------------------------------------------
// Kernel 1: fully fused window attention.
// ---------------------------------------------------------------------------
__global__ __launch_bounds__(256, 1) void fused_window_attn(
    const float* __restrict__ x,
    const float* __restrict__ bq, const float* __restrict__ bk,
    const float* __restrict__ bv, const float* __restrict__ bo,
    const float* __restrict__ conv_w, const float* __restrict__ conv_b,
    const bf16_t* __restrict__ Wqb, const bf16_t* __restrict__ Wkb,
    const bf16_t* __restrict__ Wvb, const bf16_t* __restrict__ Wob,
    float* __restrict__ out) {
  extern __shared__ char smem[];
  bf16_t* Qs = (bf16_t*)(smem + LDS_Q);
  bf16_t* Ks = (bf16_t*)(smem + LDS_K);
  bf16_t* Vt = (bf16_t*)(smem + LDS_VT);
  bf16_t* Xs = (bf16_t*)(smem + LDS_S);

  const int n    = blockIdx.x;
  const int tid  = threadIdx.x;
  const int wv   = tid >> 5;     // wave id == head id
  const int lane = tid & 31;

  // ---- Phase 0: stage X window (64x512 f32 -> bf16 LDS), coalesced -------
  // Also prefetch this wave's weight rows (L2-resident) toward the WGP.
  {
    const int row = wv * DK + lane;                  // 32 of this head's rows
    __builtin_prefetch(Wqb + row * DMODEL, 0, 0);
    __builtin_prefetch(Wkb + row * DMODEL, 0, 0);
    __builtin_prefetch(Wvb + row * DMODEL, 0, 0);
    __builtin_prefetch(Wob + row * DMODEL, 0, 0);
  }
  const float* xw = x + (size_t)n * (SL * DMODEL);
#pragma unroll 4
  for (int i = 0; i < 128; ++i) {
    const int idx = i * 256 + tid;
    Xs[idx] = (bf16_t)xw[idx];
  }
  __syncthreads();

  // ---- Phase 1: Q/K/V projections, wave w computes columns of head w -----
  // kk-outer / 16 live accumulators: each X A-fragment is loaded from LDS
  // exactly once per k-step (4x fewer ds_load than nt-outer ordering).
#pragma unroll
  for (int p = 0; p < 3; ++p) {
    const bf16_t* Wm = (p == 0) ? Wqb : (p == 1) ? Wkb : Wvb;
    const float*  bb = (p == 0) ? bq  : (p == 1) ? bk  : bv;
    v8f acc[4][4] = {};                              // [nt][mt]
    for (int kk = 0; kk < 16; ++kk) {                // K = 512 in steps of 32
      const int k0 = kk * 32;
      v16bf afr[4];
#pragma unroll
      for (int mt = 0; mt < 4; ++mt)
        afr[mt] = load_afrag(Xs, DMODEL, mt * 16, k0, lane);
#pragma unroll
      for (int nt = 0; nt < 4; ++nt) {
        const v16bf bfr = load_bfrag(Wm, DMODEL, wv * DK + nt * 16, k0, lane);
#pragma unroll
        for (int mt = 0; mt < 4; ++mt)
          acc[nt][mt] = wmma_bf16(afr[mt], bfr, acc[nt][mt]);
      }
    }
#pragma unroll
    for (int nt = 0; nt < 4; ++nt) {
      const int colbase = wv * DK + nt * 16;
      const float bval = bb[colbase + (lane & 15)];
#pragma unroll
      for (int mt = 0; mt < 4; ++mt) {
#pragma unroll
        for (int r = 0; r < 8; ++r) {
          const int Mr = mt * 16 + r + ((lane >> 4) << 3);
          const int Nc = nt * 16 + (lane & 15);
          const bf16_t hv = (bf16_t)(acc[nt][mt][r] + bval);
          if (p == 0)      Qs[wv * 4096 + Mr * 64 + Nc] = hv;   // [h][tok][dk]
          else if (p == 1) Ks[wv * 4096 + Mr * 64 + Nc] = hv;   // [h][tok][dk]
          else             Vt[wv * 4096 + Nc * 64 + Mr] = hv;   // [h][dk][tok]
        }
      }
    }
  }
  __syncthreads();   // X region is about to be reused for scores

  // ---- Phase 2a: scores = (q @ k^T) * 1/sqrt(dk), f32 to wave-private LDS
  float* Sb = (float*)(smem + LDS_S + wv * 16384);
  {
    const bf16_t* Qh = Qs + wv * 4096;
    const bf16_t* Kh = Ks + wv * 4096;
    v16bf qf[4][2];
#pragma unroll
    for (int mt = 0; mt < 4; ++mt)
#pragma unroll
      for (int kt = 0; kt < 2; ++kt)
        qf[mt][kt] = load_afrag(Qh, DK, mt * 16, kt * 32, lane);
#pragma unroll
    for (int nt = 0; nt < 4; ++nt) {
      v8f acc[4] = {};
#pragma unroll
      for (int kt = 0; kt < 2; ++kt) {
        const v16bf bfr = load_bfrag(Kh, DK, nt * 16, kt * 32, lane);
#pragma unroll
        for (int mt = 0; mt < 4; ++mt)
          acc[mt] = wmma_bf16(qf[mt][kt], bfr, acc[mt]);
      }
#pragma unroll
      for (int mt = 0; mt < 4; ++mt)
#pragma unroll
        for (int r = 0; r < 8; ++r) {
          const int Mr = mt * 16 + r + ((lane >> 4) << 3);
          const int Nc = nt * 16 + (lane & 15);
          Sb[Mr * 64 + Nc] = acc[mt][r] * 0.125f;
        }
    }
  }
  __syncthreads();   // fence f32 score stores before mixed-type reuse

  // ---- Phase 2b: 7-tap conv over the flat 4096 row + softmax per token ---
  // attn (bf16) overwrites Sb in place: attn row qi covers f32 elements
  // [qi*32, qi*32+32), which is always strictly below every conv read
  // (>= qi*64+61) of later rows; current-row reads precede the store -> hazard-free in order.
  {
    float cw[7];
#pragma unroll
    for (int t = 0; t < 7; ++t) cw[t] = conv_w[t];
    const float cbv = conv_b[0];
    bf16_t* attn = (bf16_t*)Sb;
    for (int qi = 0; qi < 64; ++qi) {            // do NOT unroll (hazard proof)
      float z[2];
#pragma unroll
      for (int e2 = 0; e2 < 2; ++e2) {
        const int s = qi * 64 + lane * 2 + e2;
        float a = Sb[s] + cbv;
#pragma unroll
        for (int t = 0; t < 7; ++t) {
          const int ss = s + t - 3;              // flat conv, zero at ends only
          const float vtap = (ss >= 0 && ss < 4096) ? Sb[ss] : 0.0f;
          a += cw[t] * vtap;
        }
        z[e2] = a;
      }
      float rmax = fmaxf(z[0], z[1]);
#pragma unroll
      for (int off = 16; off > 0; off >>= 1)
        rmax = fmaxf(rmax, __shfl_xor(rmax, off, 32));
      const float e0 = __expf(z[0] - rmax);
      const float e1 = __expf(z[1] - rmax);
      float rsum = e0 + e1;
#pragma unroll
      for (int off = 16; off > 0; off >>= 1)
        rsum += __shfl_xor(rsum, off, 32);
      const float inv = __frcp_rn(rsum);
      v2bf pk;
      pk[0] = (bf16_t)(e0 * inv);
      pk[1] = (bf16_t)(e1 * inv);
      *(v2bf*)(attn + qi * 64 + lane * 2) = pk;  // one 32-bit LDS store
    }
  }

  // ---- Phase 2c: O_h = attn @ v_h  (bf16 WMMA, B from transposed V) ------
  {
    const bf16_t* attn = (const bf16_t*)Sb;
    const bf16_t* Vh = Vt + wv * 4096;
    bf16_t* Ob = (bf16_t*)((char*)Sb + 8192);    // second half of wave buffer
    v16bf af[4][2];
#pragma unroll
    for (int mt = 0; mt < 4; ++mt)
#pragma unroll
      for (int kt = 0; kt < 2; ++kt)
        af[mt][kt] = load_afrag(attn, SL, mt * 16, kt * 32, lane);
#pragma unroll
    for (int nt = 0; nt < 4; ++nt) {
      v8f acc[4] = {};
#pragma unroll
      for (int kt = 0; kt < 2; ++kt) {
        const v16bf bfr = load_bfrag(Vh, SL, nt * 16, kt * 32, lane);
#pragma unroll
        for (int mt = 0; mt < 4; ++mt)
          acc[mt] = wmma_bf16(af[mt][kt], bfr, acc[mt]);
      }
#pragma unroll
      for (int mt = 0; mt < 4; ++mt)
#pragma unroll
        for (int r = 0; r < 8; ++r) {
          const int Mr = mt * 16 + r + ((lane >> 4) << 3);
          Ob[Mr * 64 + nt * 16 + (lane & 15)] = (bf16_t)acc[mt][r];
        }
    }
  }
  __syncthreads();   // final projection mixes all heads' O

  // ---- Phase 3: out = O @ Wo^T + bo, f32 store to HBM --------------------
  // Same kk-outer / 16-accumulator structure as phase 1.
  {
    float* outw = out + (size_t)n * (SL * DMODEL);
    v8f acc[4][4] = {};                              // [nt][mt]
    for (int kk = 0; kk < 16; ++kk) {
      const int k0 = kk * 32;
      const int hh = kk >> 1;                        // head slice of O
      const int koff = (kk & 1) * 32;
      const bf16_t* Oh = (const bf16_t*)(smem + LDS_S + hh * 16384 + 8192);
      v16bf afr[4];
#pragma unroll
      for (int mt = 0; mt < 4; ++mt)
        afr[mt] = load_afrag(Oh, DK, mt * 16, koff, lane);
#pragma unroll
      for (int nt = 0; nt < 4; ++nt) {
        const v16bf bfr = load_bfrag(Wob, DMODEL, wv * DK + nt * 16, k0, lane);
#pragma unroll
        for (int mt = 0; mt < 4; ++mt)
          acc[nt][mt] = wmma_bf16(afr[mt], bfr, acc[nt][mt]);
      }
    }
#pragma unroll
    for (int nt = 0; nt < 4; ++nt) {
      const int colbase = wv * DK + nt * 16;
      const float bval = bo[colbase + (lane & 15)];
#pragma unroll
      for (int mt = 0; mt < 4; ++mt)
#pragma unroll
        for (int r = 0; r < 8; ++r) {
          const int Mr = mt * 16 + r + ((lane >> 4) << 3);
          outw[Mr * DMODEL + colbase + (lane & 15)] = acc[nt][mt][r] + bval;
        }
    }
  }
}

// ---------------------------------------------------------------------------
extern "C" void kernel_launch(void* const* d_in, const int* in_sizes, int n_in,
                              void* d_out, int out_size, void* d_ws,
                              size_t ws_size, hipStream_t stream) {
  const float* x  = (const float*)d_in[0];
  const float* Wq = (const float*)d_in[1];
  const float* bq = (const float*)d_in[2];
  const float* Wk = (const float*)d_in[3];
  const float* bk = (const float*)d_in[4];
  const float* Wv = (const float*)d_in[5];
  const float* bv = (const float*)d_in[6];
  const float* Wo = (const float*)d_in[7];
  const float* bo = (const float*)d_in[8];
  const float* cw = (const float*)d_in[9];
  const float* cb = (const float*)d_in[10];

  bf16_t* wsb = (bf16_t*)d_ws;                       // 4 * 512*512 bf16 = 2 MB
  const int WSZ = 512 * 512;

  hipLaunchKernelGGL(weights_to_bf16, dim3((4 * WSZ) / 256), dim3(256), 0,
                     stream, Wq, Wk, Wv, Wo, wsb);

  (void)hipFuncSetAttribute((const void*)fused_window_attn,
                            hipFuncAttributeMaxDynamicSharedMemorySize,
                            LDS_BYTES);
  hipLaunchKernelGGL(fused_window_attn, dim3(NWIN), dim3(256), LDS_BYTES,
                     stream, x, bq, bk, bv, bo, cw, cb,
                     wsb, wsb + WSZ, wsb + 2 * WSZ, wsb + 3 * WSZ,
                     (float*)d_out);
}
